// GATv2Classifier_764504179505
// MI455X (gfx1250) — matrix operand order repaired and checked by reading, not compiled, not run on hardware
//
#include <hip/hip_runtime.h>
#include <hip/hip_bf16.h>

#define NN 50000
#define NE 600000
#define NE2 (NE + NN)
#define HC 128
#define NH 4
#define NG 32
#define ODIM 10
#define NEG 0.2f

typedef __attribute__((ext_vector_type(16))) __bf16 v16bf;
typedef __attribute__((ext_vector_type(8)))  float v8f;
typedef unsigned int u32x4 __attribute__((ext_vector_type(4)));
typedef int          i32x8 __attribute__((ext_vector_type(8)));
typedef int          i32x4 __attribute__((ext_vector_type(4)));

// ---- order-preserving float <-> uint keys for atomic segment-max ----
__device__ __forceinline__ unsigned fkey(float f) {
    unsigned u = __float_as_uint(f);
    return (u & 0x80000000u) ? ~u : (u | 0x80000000u);
}
__device__ __forceinline__ float funkey(unsigned k) {
    return (k & 0x80000000u) ? __uint_as_float(k ^ 0x80000000u)
                             : __uint_as_float(~k);
}

// ---------------- fills ----------------
__global__ void fill_f32(float* p, float v, int n) {
    int i = blockIdx.x * blockDim.x + threadIdx.x;
    if (i < n) p[i] = v;
}
__global__ void fill_u32(unsigned* p, unsigned v, int n) {
    int i = blockIdx.x * blockDim.x + threadIdx.x;
    if (i < n) p[i] = v;
}

// ---------------- self-loop attr (PyG fill_value='mean') ----------------
__global__ void loop_accum(const int* __restrict__ ei, const float* __restrict__ ew,
                           float* __restrict__ sumw, float* __restrict__ cntf) {
    int e = blockIdx.x * blockDim.x + threadIdx.x;
    if (e >= NE) return;
    int dst = ei[NE + e];
    atomicAdd(&sumw[dst], ew[e]);
    atomicAdd(&cntf[dst], 1.0f);
}
__global__ void loop_final(float* __restrict__ sumw, const float* __restrict__ cntf) {
    int i = blockIdx.x * blockDim.x + threadIdx.x;
    if (i < NN) sumw[i] = sumw[i] / fmaxf(cntf[i], 1.0f);
}

// ---------------- bf16 WMMA GEMM: Y[M,128] = X[M,128] @ W[128,128] + b ----------------
// 4 waves/block; each wave computes a 16-row tile.
// Weight staging: TDM (tensor_load_to_lds) DMAs W (128x128 f32, 64KB) into LDS,
// then the block swizzles it to bf16 WMMA B-fragment layout so the inner loop
// reads each 16-element fragment as one contiguous 32B LDS access.
__global__ __launch_bounds__(128) void gemm_wmma(const float* __restrict__ X,
                                                 const float* __restrict__ W,
                                                 const float* __restrict__ b,
                                                 float* __restrict__ Y, int nrows) {
    __shared__ float  sStage[HC * HC];        // 64 KB raw f32 weights (TDM dest)
    __shared__ v16bf  sFrag[8 * 4 * 32];      // 32 KB bf16 B fragments [t][s][lane]
    __shared__ float  sB[HC];

    // --- TDM: wave 0 issues one 2D tensor DMA of W into sStage ---
    if (threadIdx.x < 32) {
        const unsigned lds_off = (unsigned)(size_t)(void*)sStage;
        const unsigned long long ga = (unsigned long long)(size_t)W;
        // D# group0: count=1 | lds_addr | global_addr[56:0] | type=2
        u32x4 g0 = { 1u,
                     lds_off,
                     (unsigned)(ga & 0xFFFFFFFFu),
                     (unsigned)((ga >> 32) & 0x01FFFFFFu) | (2u << 30) };
        // D# group1: data_size=2 (4B); tensor_dim0=128 (bits 79:48);
        // tensor_dim1=128 (111:80); tile_dim0=128 (127:112); tile_dim1=128 (143:128);
        // tensor_dim0_stride=128 (207:160); tensor_dim1_stride=16384 (255:208)
        i32x8 g1 = { (int)(2u << 16),
                     (int)(128u << 16),
                     (int)(128u << 16),
                     (int)(128u << 16),
                     128,
                     128,
                     (int)(16384u << 16),
                     0 };
        i32x4 gz4 = { 0, 0, 0, 0 };
        i32x8 gz8 = { 0, 0, 0, 0, 0, 0, 0, 0 };
        __builtin_amdgcn_tensor_load_to_lds(g0, g1, gz4, gz4, gz8, 0);
        __builtin_amdgcn_s_wait_tensorcnt(0);
    }
    if (threadIdx.x < HC) sB[threadIdx.x] = b[threadIdx.x];
    __syncthreads();

    // --- swizzle f32 staging -> bf16 B-fragment layout ---
    // element (k,n): s=k>>5, half=(k>>3)&1, i=(k&7)+((k>>4)&1)*8, t=n>>4, lo=n&15
    // dest: sFrag[(t*4+s)*32 + half*16 + lo][i]
    {
        __bf16* frag = (__bf16*)sFrag;
        for (int f = threadIdx.x; f < HC * HC; f += 128) {
            const int k = f >> 7, n = f & 127;
            const int t = n >> 4, lo = n & 15;
            const int s = k >> 5, half = (k >> 3) & 1, i = (k & 7) + (((k >> 4) & 1) << 3);
            frag[((t * 4 + s) * 32 + half * 16 + lo) * 16 + i] = (__bf16)sStage[f];
        }
    }
    __syncthreads();

    const int wave = threadIdx.x >> 5;
    const int lane = threadIdx.x & 31;
    const int tile = blockIdx.x * 4 + wave;     // 16-row tile index
    const int m0   = tile * 16;
    if (m0 >= nrows) return;                    // uniform per wave; nrows % 16 == 0

    const int half = lane >> 4;                 // 0: lanes 0-15, 1: lanes 16-31
    const int mrow = m0 + (lane & 15);
    const float4* xv = (const float4*)(X + (size_t)mrow * HC);

    // A fragments per ISA 16-bit 16x32 layout:
    //   i<8  : k = 32s + 8*half + i
    //   i>=8 : k = 32s + 8*half + 16 + (i-8)
    v16bf a[4];
#pragma unroll
    for (int s = 0; s < 4; ++s) {
        const int kb = s * 32 + half * 8;       // multiple of 8
        const float4 p0 = xv[kb >> 2];
        const float4 p1 = xv[(kb >> 2) + 1];
        const float4 q0 = xv[(kb + 16) >> 2];
        const float4 q1 = xv[((kb + 16) >> 2) + 1];
        a[s][0]  = (__bf16)p0.x;  a[s][1]  = (__bf16)p0.y;
        a[s][2]  = (__bf16)p0.z;  a[s][3]  = (__bf16)p0.w;
        a[s][4]  = (__bf16)p1.x;  a[s][5]  = (__bf16)p1.y;
        a[s][6]  = (__bf16)p1.z;  a[s][7]  = (__bf16)p1.w;
        a[s][8]  = (__bf16)q0.x;  a[s][9]  = (__bf16)q0.y;
        a[s][10] = (__bf16)q0.z;  a[s][11] = (__bf16)q0.w;
        a[s][12] = (__bf16)q1.x;  a[s][13] = (__bf16)q1.y;
        a[s][14] = (__bf16)q1.z;  a[s][15] = (__bf16)q1.w;
    }

    const int ncol = lane & 15;
#pragma unroll
    for (int t = 0; t < 8; ++t) {
        const int n = t * 16 + ncol;
        v8f acc = {};
#pragma unroll
        for (int s = 0; s < 4; ++s) {
            const v16bf bf = sFrag[(t * 4 + s) * 32 + lane];   // one 32B LDS read
            acc = __builtin_amdgcn_wmma_f32_16x16x32_bf16(
                false, a[s], false, bf, (short)0, acc, false, false);
        }
        const float bias = sB[n];
        // C/D layout: VGPR r -> row m0 + r + 8*half, col n
#pragma unroll
        for (int r = 0; r < 8; ++r)
            Y[(size_t)(m0 + r + half * 8) * HC + n] = acc[r] + bias;
    }
}

// ---------------- per-edge attention logits + segment max ----------------
// one wave per edge; lane l, iteration j handles channel c = l + 32*j (head j)
__global__ void alpha_kernel(const int* __restrict__ ei, const float* __restrict__ ew,
                             const float* __restrict__ loopw,
                             const float* __restrict__ XL, const float* __restrict__ XR,
                             const float* __restrict__ We, const float* __restrict__ att,
                             float* __restrict__ AL, unsigned* __restrict__ AMAX) {
    const int e = blockIdx.x * 8 + (threadIdx.x >> 5);
    if (e >= NE2) return;
    const int lane = threadIdx.x & 31;
    int src, dst; float w;
    if (e < NE) { src = ei[e]; dst = ei[NE + e]; w = ew[e]; }
    else        { src = dst = e - NE;            w = loopw[src]; }
    const float* xl = XL + (size_t)src * HC;
    const float* xr = XR + (size_t)dst * HC;
#pragma unroll
    for (int j = 0; j < NH; ++j) {
        const int c = lane + 32 * j;
        float m = xl[c] + xr[c] + w * We[c];
        m = (m > 0.0f) ? m : NEG * m;          // leaky_relu
        float p = m * att[c];
#pragma unroll
        for (int off = 16; off > 0; off >>= 1) p += __shfl_xor(p, off, 32);
        if (lane == 0) {
            AL[(size_t)e * NH + j] = p;
            atomicMax(&AMAX[(size_t)dst * NH + j], fkey(p));
        }
    }
}

// ---------------- exp(alpha - max) + segment sum (in-place over AL) ----------------
__global__ void ex_kernel(const int* __restrict__ ei, float* __restrict__ AL,
                          const unsigned* __restrict__ AMAX, float* __restrict__ DEN) {
    const long idx = (long)blockIdx.x * blockDim.x + threadIdx.x;
    if (idx >= (long)NE2 * NH) return;
    const int e = (int)(idx >> 2), j = (int)(idx & 3);
    const int dst = (e < NE) ? ei[NE + e] : (e - NE);
    const float ev = expf(AL[idx] - funkey(AMAX[(size_t)dst * NH + j]));
    AL[idx] = ev;
    atomicAdd(&DEN[(size_t)dst * NH + j], ev);
}

// ---------------- weighted aggregation: ACC[dst] += (ex/denom) * XL[src] ----------------
__global__ void agg_kernel(const int* __restrict__ ei, const float* __restrict__ XL,
                           const float* __restrict__ AL, const float* __restrict__ DEN,
                           float* __restrict__ ACC) {
    const int e = blockIdx.x * 8 + (threadIdx.x >> 5);
    if (e >= NE2) return;
    const int lane = threadIdx.x & 31;
    int src, dst;
    if (e < NE) { src = ei[e]; dst = ei[NE + e]; }
    else        { src = dst = e - NE; }
    const float* xl = XL + (size_t)src * HC;
#pragma unroll
    for (int j = 0; j < NH; ++j) {
        const int c = lane + 32 * j;
        const float a = AL[(size_t)e * NH + j] /
                        (DEN[(size_t)dst * NH + j] + 1e-16f);
        atomicAdd(&ACC[(size_t)dst * HC + c], a * xl[c]);
    }
}

__global__ void addbias_kernel(const float* __restrict__ ACC, const float* __restrict__ bias,
                               float* __restrict__ Hh) {
    const long idx = (long)blockIdx.x * blockDim.x + threadIdx.x;
    if (idx >= (long)NN * HC) return;
    Hh[idx] = ACC[idx] + bias[idx & (HC - 1)];
}

// ---------------- pooling + FC ----------------
__global__ void pool_cnt(const int* __restrict__ batch, float* __restrict__ PCNT) {
    int i = blockIdx.x * blockDim.x + threadIdx.x;
    if (i < NN) atomicAdd(&PCNT[batch[i]], 1.0f);
}
__global__ void pool_sum(const int* __restrict__ batch, const float* __restrict__ Hh,
                         float* __restrict__ POOL) {
    const long idx = (long)blockIdx.x * blockDim.x + threadIdx.x;
    if (idx >= (long)NN * HC) return;
    const int i = (int)(idx >> 7), c = (int)(idx & (HC - 1));
    atomicAdd(&POOL[batch[i] * HC + c], Hh[idx]);
}
__global__ void pool_div(const float* __restrict__ POOL, const float* __restrict__ PCNT,
                         float* __restrict__ pooled_out) {
    int idx = blockIdx.x * blockDim.x + threadIdx.x;
    if (idx >= NG * HC) return;
    pooled_out[idx] = POOL[idx] / fmaxf(PCNT[idx >> 7], 1.0f);
}
__global__ void fc_kernel(const float* __restrict__ pooled, const float* __restrict__ fcW,
                          const float* __restrict__ fcb, float* __restrict__ out) {
    int t = threadIdx.x;                 // 0..319
    if (t >= NG * ODIM) return;
    const int g = t / ODIM, o = t % ODIM;
    float s = fcb[o];
    for (int c = 0; c < HC; ++c) s += pooled[g * HC + c] * fcW[c * ODIM + o];
    out[g * ODIM + o] = s;
}

extern "C" void kernel_launch(void* const* d_in, const int* in_sizes, int n_in,
                              void* d_out, int out_size, void* d_ws, size_t ws_size,
                              hipStream_t stream) {
    (void)in_sizes; (void)n_in; (void)out_size; (void)ws_size;
    const float* x      = (const float*)d_in[0];
    const int*   ei     = (const int*)d_in[1];
    const float* ew     = (const float*)d_in[2];
    const int*   batch  = (const int*)d_in[3];
    const float* p1Wl   = (const float*)d_in[4];
    const float* p1bl   = (const float*)d_in[5];
    const float* p1Wr   = (const float*)d_in[6];
    const float* p1br   = (const float*)d_in[7];
    const float* p1We   = (const float*)d_in[8];
    const float* p1att  = (const float*)d_in[9];
    const float* p1bias = (const float*)d_in[10];
    const float* p2Wl   = (const float*)d_in[11];
    const float* p2bl   = (const float*)d_in[12];
    const float* p2Wr   = (const float*)d_in[13];
    const float* p2br   = (const float*)d_in[14];
    const float* p2We   = (const float*)d_in[15];
    const float* p2att  = (const float*)d_in[16];
    const float* p2bias = (const float*)d_in[17];
    const float* fcW    = (const float*)d_in[18];
    const float* fcb    = (const float*)d_in[19];
    float* out = (float*)d_out;

    float* base = (float*)d_ws;
    float* XL   = base; base += (size_t)NN * HC;
    float* XR   = base; base += (size_t)NN * HC;
    float* ACC  = base; base += (size_t)NN * HC;
    float* Hh   = base; base += (size_t)NN * HC;
    float* AL   = base; base += (size_t)NE2 * NH;
    unsigned* AMAX = (unsigned*)base; base += (size_t)NN * NH;
    float* DEN  = base; base += (size_t)NN * NH;
    float* SUMW = base; base += NN;
    float* CNTF = base; base += NN;
    float* POOL = base; base += NG * HC;
    float* PCNT = base; base += NG;

    dim3 b256(256);
    auto cdiv = [](long a, long b) { return (unsigned)((a + b - 1) / b); };

    // self-loop mean edge attrs (SUMW/CNTF contiguous -> single fill)
    fill_f32<<<cdiv(2 * NN, 256), b256, 0, stream>>>(SUMW, 0.0f, 2 * NN);
    loop_accum<<<cdiv(NE, 256), b256, 0, stream>>>(ei, ew, SUMW, CNTF);
    loop_final<<<cdiv(NN, 256), b256, 0, stream>>>(SUMW, CNTF);

    const unsigned gemm_blocks = cdiv(NN, 64);       // 4 x 16-row tiles per block
    const unsigned edge_blocks = cdiv(NE2, 8);       // 8 edges (waves) per block

    auto layer = [&](const float* in, const float* Wl, const float* bl,
                     const float* Wr, const float* br, const float* We,
                     const float* att, const float* bias) {
        gemm_wmma<<<gemm_blocks, 128, 0, stream>>>(in, Wl, bl, XL, NN);
        gemm_wmma<<<gemm_blocks, 128, 0, stream>>>(in, Wr, br, XR, NN);
        fill_u32<<<cdiv(NN * NH, 256), b256, 0, stream>>>(AMAX, 0x007FFFFFu, NN * NH); // key(-inf)
        fill_f32<<<cdiv(NN * NH, 256), b256, 0, stream>>>(DEN, 0.0f, NN * NH);
        fill_f32<<<cdiv((long)NN * HC, 256), b256, 0, stream>>>(ACC, 0.0f, NN * HC);
        alpha_kernel<<<edge_blocks, b256, 0, stream>>>(ei, ew, SUMW, XL, XR, We, att, AL, AMAX);
        ex_kernel<<<cdiv((long)NE2 * NH, 256), b256, 0, stream>>>(ei, AL, AMAX, DEN);
        agg_kernel<<<edge_blocks, b256, 0, stream>>>(ei, XL, AL, DEN, ACC);
        addbias_kernel<<<cdiv((long)NN * HC, 256), b256, 0, stream>>>(ACC, bias, Hh);
    };

    layer(x,  p1Wl, p1bl, p1Wr, p1br, p1We, p1att, p1bias);
    layer(Hh, p2Wl, p2bl, p2Wr, p2br, p2We, p2att, p2bias);

    // mean pool per graph + FC; output layout: out[32*10] then pooled[32*128]
    fill_f32<<<cdiv(NG * HC + NG, 256), b256, 0, stream>>>(POOL, 0.0f, NG * HC + NG);
    pool_cnt<<<cdiv(NN, 256), b256, 0, stream>>>(batch, PCNT);
    pool_sum<<<cdiv((long)NN * HC, 256), b256, 0, stream>>>(batch, Hh, POOL);
    pool_div<<<cdiv(NG * HC, 256), b256, 0, stream>>>(POOL, PCNT, out + NG * ODIM);
    fc_kernel<<<1, NG * ODIM, 0, stream>>>(out + NG * ODIM, fcW, fcb, out);
}
